// NeuronRouter_22282290331738
// MI455X (gfx1250) — matrix operand order repaired and checked by reading, not compiled, not run on hardware
//
#include <hip/hip_runtime.h>
#include <hip/hip_bf16.h>

// ---------------------------------------------------------------------------
// NeuronRouter for MI455X (gfx1250): bf16 WMMA (16x16x32, f32 accum) for all
// GEMM-like stages; scoring GEMM algebraically fused: w0*x@N^T + w1*ctx@N^T
// == (w0*x + w1*ctx)@N^T. Round 2: register-blocked GEMMs (4 WMMA / K-step
// per wave) to double fragment-level arithmetic intensity.
// ---------------------------------------------------------------------------

#define B_  4
#define S_  1024
#define D_  1024
#define H_  16
#define DH_ 64
#define N_  16384
#define K_  16
#define M_  (B_ * S_)   // 4096 tokens

typedef __attribute__((ext_vector_type(16))) __bf16 v16bf;
typedef __attribute__((ext_vector_type(8)))  float  v8f;

union FragB16 { v16bf v; uint4 q[2]; };

__device__ __forceinline__ unsigned short f2bf(float f) {
    unsigned int u = __float_as_uint(f);
    u += 0x7fffu + ((u >> 16) & 1u);        // round-to-nearest-even
    return (unsigned short)(u >> 16);
}

__device__ __forceinline__ v8f wmma_bf16(const FragB16& a, const FragB16& b, v8f c) {
    // (neg_a, A, neg_b, B, c_mod, C, reuse_a, reuse_b)
    return __builtin_amdgcn_wmma_f32_16x16x32_bf16(false, a.v, false, b.v,
                                                   (short)0, c, false, false);
}

// ---------------------------------------------------------------------------
// fp32 -> bf16 (grid-stride)
// ---------------------------------------------------------------------------
__global__ void cvt_bf16_k(const float* __restrict__ src,
                           unsigned short* __restrict__ dst, int n) {
    int i = blockIdx.x * blockDim.x + threadIdx.x;
    int stride = gridDim.x * blockDim.x;
    for (; i < n; i += stride) dst[i] = f2bf(src[i]);
}

// fp32 [rows,cols] -> bf16 transposed [cols,rows]
__global__ void cvt_tr_bf16_k(const float* __restrict__ src,
                              unsigned short* __restrict__ dst,
                              int rows, int cols) {
    int i = blockIdx.x * blockDim.x + threadIdx.x;
    int stride = gridDim.x * blockDim.x;
    int total = rows * cols;
    for (; i < total; i += stride) {
        int r = i / cols, c = i % cols;
        dst[(size_t)c * rows + r] = f2bf(src[i]);
    }
}

__global__ void zero_k(float* __restrict__ p, size_t n) {
    size_t i = (size_t)blockIdx.x * blockDim.x + threadIdx.x;
    size_t stride = (size_t)gridDim.x * blockDim.x;
    for (; i < n; i += stride) p[i] = 0.0f;
}

// ---------------------------------------------------------------------------
// QKV GEMM: C[M,1024] = Xbf[M,1024] * Wt[1024,1024]^T + bias
// Wt is row-major [Nout, K] (pre-transposed) so B-fragments load contiguously.
// One wave computes a 32x32 C tile (2 A frags x 2 B frags -> 4 WMMA/K-step).
// Block = 256 threads = 8 waves tiling 128(M) x 64(N).
// vmode=0: out[b,h,s,dh]; vmode=1: out[b,h,dh,s]
// grid = (Nout/64, M/128)
// ---------------------------------------------------------------------------
__global__ __launch_bounds__(256)
void qkv_gemm_k(const unsigned short* __restrict__ xbf,
                const unsigned short* __restrict__ wt,
                const float* __restrict__ bias,
                unsigned short* __restrict__ out, int vmode) {
    int wave = threadIdx.x >> 5;
    int lane = threadIdx.x & 31;
    int l16  = lane & 15;
    int kh   = lane >> 4;
    int wm   = wave & 3;          // 4 waves along M
    int wn   = wave >> 2;         // 2 waves along N

    int tileM = blockIdx.y * 128 + wm * 32;
    int tileN = blockIdx.x * 64  + wn * 32;

    const unsigned short* arow0 = xbf + (size_t)(tileM + l16) * D_;
    const unsigned short* arow1 = arow0 + (size_t)16 * D_;
    const unsigned short* brow0 = wt  + (size_t)(tileN + l16) * D_;
    const unsigned short* brow1 = brow0 + (size_t)16 * D_;

    v8f acc00 = {}, acc01 = {}, acc10 = {}, acc11 = {};
    for (int k0 = 0; k0 < D_; k0 += 32) {
        FragB16 a0, a1, b0, b1;
        a0.q[0] = *(const uint4*)(arow0 + k0 + kh * 16);
        a0.q[1] = *(const uint4*)(arow0 + k0 + kh * 16 + 8);
        a1.q[0] = *(const uint4*)(arow1 + k0 + kh * 16);
        a1.q[1] = *(const uint4*)(arow1 + k0 + kh * 16 + 8);
        b0.q[0] = *(const uint4*)(brow0 + k0 + kh * 16);
        b0.q[1] = *(const uint4*)(brow0 + k0 + kh * 16 + 8);
        b1.q[0] = *(const uint4*)(brow1 + k0 + kh * 16);
        b1.q[1] = *(const uint4*)(brow1 + k0 + kh * 16 + 8);
        acc00 = wmma_bf16(a0, b0, acc00);
        acc01 = wmma_bf16(a0, b1, acc01);
        acc10 = wmma_bf16(a1, b0, acc10);
        acc11 = wmma_bf16(a1, b1, acc11);
    }

    auto store_tile = [&](v8f acc, int rbase, int ncol) {
        float bv = bias[ncol];
        int h = ncol >> 6, dh = ncol & 63;
#pragma unroll
        for (int v = 0; v < 8; ++v) {
            int r  = rbase + v + kh * 8;        // C row = v + khalf*8
            int bb = r >> 10, s = r & (S_ - 1);
            float val = acc[v] + bv;
            size_t o;
            if (vmode) o = ((size_t)(bb * H_ + h) * DH_ + dh) * S_ + s;   // v^T
            else       o = ((size_t)(bb * H_ + h) * S_  + s ) * DH_ + dh; // q/k
            out[o] = f2bf(val);
        }
    };
    store_tile(acc00, tileM,      tileN + l16);
    store_tile(acc01, tileM,      tileN + 16 + l16);
    store_tile(acc10, tileM + 16, tileN + l16);
    store_tile(acc11, tileM + 16, tileN + 16 + l16);
}

// ---------------------------------------------------------------------------
// Flash-style attention: one wave handles one 16-row q tile of one (b,h).
// Online softmax; P re-shaped C-layout -> A-layout via per-wave LDS tile.
// grid = 512, block = 256 (8 waves) -> 4096 waves = B*H*(S/16)
// ---------------------------------------------------------------------------
__global__ __launch_bounds__(256)
void attn_k(const unsigned short* __restrict__ qbf,   // [B,H,S,DH] bf16
            const unsigned short* __restrict__ kbf,   // [B,H,S,DH] bf16
            const unsigned short* __restrict__ vtbf,  // [B,H,DH,S] bf16
            float* __restrict__ ctx) {                // [B,S,D] f32
    __shared__ __align__(16) unsigned short pt[8][16][32];

    int wave = threadIdx.x >> 5;
    int lane = threadIdx.x & 31;
    int l16  = lane & 15;
    int kh   = lane >> 4;

    int wid = blockIdx.x * 8 + wave;      // 0..4095
    int qt  = wid & 63;                   // q tile within head
    int bh  = wid >> 6;                   // b*H + h
    int h   = bh & (H_ - 1);
    int b   = bh >> 4;
    int q0  = qt * 16;

    // A-fragments of Q (16x64 -> two 16x32 frags), kept live across the loop
    const unsigned short* qbase = qbf + ((size_t)bh * S_ + q0 + l16) * DH_;
    FragB16 qa0, qa1;
    qa0.q[0] = *(const uint4*)(qbase + kh * 16);
    qa0.q[1] = *(const uint4*)(qbase + kh * 16 + 8);
    qa1.q[0] = *(const uint4*)(qbase + 32 + kh * 16);
    qa1.q[1] = *(const uint4*)(qbase + 32 + kh * 16 + 8);

    v8f o0 = {}, o1 = {}, o2 = {}, o3 = {};
    float rmax[8], rsum[8];
#pragma unroll
    for (int v = 0; v < 8; ++v) { rmax[v] = -INFINITY; rsum[v] = 0.0f; }

    const float scale = 0.125f;   // 1/sqrt(64)

    for (int kc0 = 0; kc0 < S_; kc0 += 32) {
        // ---- scores S[16 x 32] = Q_tile @ K_chunk^T ----
        v8f s0 = {}, s1 = {};
        const unsigned short* kb0 = kbf + ((size_t)bh * S_ + kc0 + l16) * DH_;
        const unsigned short* kb1 = kbf + ((size_t)bh * S_ + kc0 + 16 + l16) * DH_;
        FragB16 f;
        f.q[0] = *(const uint4*)(kb0 + kh * 16);
        f.q[1] = *(const uint4*)(kb0 + kh * 16 + 8);
        s0 = wmma_bf16(qa0, f, s0);
        f.q[0] = *(const uint4*)(kb0 + 32 + kh * 16);
        f.q[1] = *(const uint4*)(kb0 + 32 + kh * 16 + 8);
        s0 = wmma_bf16(qa1, f, s0);
        f.q[0] = *(const uint4*)(kb1 + kh * 16);
        f.q[1] = *(const uint4*)(kb1 + kh * 16 + 8);
        s1 = wmma_bf16(qa0, f, s1);
        f.q[0] = *(const uint4*)(kb1 + 32 + kh * 16);
        f.q[1] = *(const uint4*)(kb1 + 32 + kh * 16 + 8);
        s1 = wmma_bf16(qa1, f, s1);

        // ---- online softmax (row stats across the 16-lane column groups) ----
#pragma unroll
        for (int v = 0; v < 8; ++v) {
            float a = s0[v] * scale;
            float c = s1[v] * scale;
            float t = fmaxf(a, c);
            t = fmaxf(t, __shfl_xor(t, 1, 32));
            t = fmaxf(t, __shfl_xor(t, 2, 32));
            t = fmaxf(t, __shfl_xor(t, 4, 32));
            t = fmaxf(t, __shfl_xor(t, 8, 32));
            float nm   = fmaxf(rmax[v], t);
            float corr = __expf(rmax[v] - nm);
            float p0   = __expf(a - nm);
            float p1   = __expf(c - nm);
            float ps   = p0 + p1;
            ps += __shfl_xor(ps, 1, 32);
            ps += __shfl_xor(ps, 2, 32);
            ps += __shfl_xor(ps, 4, 32);
            ps += __shfl_xor(ps, 8, 32);
            rsum[v] = rsum[v] * corr + ps;
            rmax[v] = nm;
            o0[v] *= corr; o1[v] *= corr; o2[v] *= corr; o3[v] *= corr;
            int r = v + kh * 8;
            pt[wave][r][l16]      = f2bf(p0);
            pt[wave][r][16 + l16] = f2bf(p1);
        }
        __builtin_amdgcn_wave_barrier();
        FragB16 p;
        const unsigned short* pr = &pt[wave][l16][0];
        p.q[0] = *(const uint4*)(pr + kh * 16);
        p.q[1] = *(const uint4*)(pr + kh * 16 + 8);
        __builtin_amdgcn_wave_barrier();

        // ---- O += P @ V_chunk  (V^T rows contiguous along s_k) ----
        const unsigned short* vbase = vtbf + (size_t)bh * DH_ * S_ + kc0 + kh * 16;
#define PV_STEP(T, OREG)                                                    \
        {                                                                   \
            const unsigned short* vp = vbase + (size_t)((T)*16 + l16) * S_; \
            FragB16 vb;                                                     \
            vb.q[0] = *(const uint4*)vp;                                    \
            vb.q[1] = *(const uint4*)(vp + 8);                              \
            OREG = wmma_bf16(p, vb, OREG);                                  \
        }
        PV_STEP(0, o0) PV_STEP(1, o1) PV_STEP(2, o2) PV_STEP(3, o3)
#undef PV_STEP
    }

#pragma unroll
    for (int v = 0; v < 8; ++v) {
        float inv = 1.0f / rsum[v];
        size_t rowoff = (size_t)(b * S_ + q0 + v + kh * 8) * D_ + h * DH_;
        ctx[rowoff +  0 + l16] = o0[v] * inv;
        ctx[rowoff + 16 + l16] = o1[v] * inv;
        ctx[rowoff + 32 + l16] = o2[v] * inv;
        ctx[rowoff + 48 + l16] = o3[v] * inv;
    }
}

// ---------------------------------------------------------------------------
// Path weights + mix: m = w0*x + w1*ctx (bf16). One block per token.
// ---------------------------------------------------------------------------
__global__ __launch_bounds__(256)
void path_mix_k(const float* __restrict__ x, const float* __restrict__ ctx,
                const float* __restrict__ pw, const float* __restrict__ pb,
                unsigned short* __restrict__ mbf) {
    __shared__ float r0[256], r1[256];
    __shared__ float w0s, w1s;
    int row = blockIdx.x;
    int tid = threadIdx.x;
    const float* xr = x   + (size_t)row * D_;
    const float* cr = ctx + (size_t)row * D_;
    float a0 = 0.0f, a1 = 0.0f;
    for (int d = tid; d < D_; d += 256) {
        float xv = xr[d], cv = cr[d];
        a0 += xv * pw[2 * d]     + cv * pw[2 * (D_ + d)];
        a1 += xv * pw[2 * d + 1] + cv * pw[2 * (D_ + d) + 1];
    }
    r0[tid] = a0; r1[tid] = a1;
    __syncthreads();
    for (int s = 128; s > 0; s >>= 1) {
        if (tid < s) { r0[tid] += r0[tid + s]; r1[tid] += r1[tid + s]; }
        __syncthreads();
    }
    if (tid == 0) {
        float l0 = r0[0] + pb[0], l1 = r1[0] + pb[1];
        float mx = fmaxf(l0, l1);
        float e0 = __expf(l0 - mx), e1 = __expf(l1 - mx);
        float inv = 1.0f / (e0 + e1);
        w0s = e0 * inv; w1s = e1 * inv;
    }
    __syncthreads();
    float w0 = w0s, w1 = w1s;
    for (int d = tid; d < D_; d += 256)
        mbf[(size_t)row * D_ + d] = f2bf(w0 * xr[d] + w1 * cr[d]);
}

// ---------------------------------------------------------------------------
// Fused scoring GEMM + running top-16 + softmax + gather + scatter.
// 16 tokens per block; 8 waves, each computing 16 tokens x 64 neurons
// (A frag reused across 4 B frags -> 4 WMMA/K-step). Block tile = 16 x 512.
// Neurons (32 MB bf16) stay L2-resident across the 256 blocks.
// ---------------------------------------------------------------------------
__global__ __launch_bounds__(256)
void score_topk_k(const unsigned short* __restrict__ mbf,   // [M,D] bf16
                  const unsigned short* __restrict__ nbf,   // [N,D] bf16
                  const float* __restrict__ neurons,        // [N,D] f32
                  float* __restrict__ outv,                 // [M,D]
                  float* __restrict__ oidx,                 // [M,K] (as float)
                  float* __restrict__ ow,                   // [M,K]
                  float* __restrict__ osel) {               // [M,N] pre-zeroed
    __shared__ float sc[16][512];
    __shared__ float topV[16][K_];
    __shared__ int   topI[16][K_];
    __shared__ float tw[16][K_];

    int tid  = threadIdx.x;
    int wave = tid >> 5;
    int lane = tid & 31;
    int l16  = lane & 15;
    int kh   = lane >> 4;
    int tok0 = blockIdx.x * 16;

    float minv = -INFINITY;
    int   minslot = 0;
    if (tid < 16) {
#pragma unroll
        for (int k = 0; k < K_; ++k) { topV[tid][k] = -INFINITY; topI[tid][k] = 0; }
    }
    __syncthreads();

    const unsigned short* arow = mbf + (size_t)(tok0 + l16) * D_;

    for (int nt = 0; nt < N_ / 512; ++nt) {
        int nbase = nt * 512 + wave * 64;           // this wave's 64 neurons
        const unsigned short* br0 = nbf + (size_t)(nbase + l16) * D_;
        const unsigned short* br1 = br0 + (size_t)16 * D_;
        const unsigned short* br2 = br0 + (size_t)32 * D_;
        const unsigned short* br3 = br0 + (size_t)48 * D_;

        v8f ac0 = {}, ac1 = {}, ac2 = {}, ac3 = {};
        for (int k0 = 0; k0 < D_; k0 += 32) {
            FragB16 a, b0, b1, b2, b3;
            a.q[0]  = *(const uint4*)(arow + k0 + kh * 16);
            a.q[1]  = *(const uint4*)(arow + k0 + kh * 16 + 8);
            b0.q[0] = *(const uint4*)(br0 + k0 + kh * 16);
            b0.q[1] = *(const uint4*)(br0 + k0 + kh * 16 + 8);
            b1.q[0] = *(const uint4*)(br1 + k0 + kh * 16);
            b1.q[1] = *(const uint4*)(br1 + k0 + kh * 16 + 8);
            b2.q[0] = *(const uint4*)(br2 + k0 + kh * 16);
            b2.q[1] = *(const uint4*)(br2 + k0 + kh * 16 + 8);
            b3.q[0] = *(const uint4*)(br3 + k0 + kh * 16);
            b3.q[1] = *(const uint4*)(br3 + k0 + kh * 16 + 8);
            ac0 = wmma_bf16(a, b0, ac0);
            ac1 = wmma_bf16(a, b1, ac1);
            ac2 = wmma_bf16(a, b2, ac2);
            ac3 = wmma_bf16(a, b3, ac3);
        }
#pragma unroll
        for (int v = 0; v < 8; ++v) {
            int r = v + kh * 8;
            int c = wave * 64 + l16;
            sc[r][c]      = ac0[v];
            sc[r][c + 16] = ac1[v];
            sc[r][c + 32] = ac2[v];
            sc[r][c + 48] = ac3[v];
        }
        __syncthreads();

        if (tid < 16) {
            int nb = nt * 512;
            for (int j = 0; j < 512; ++j) {
                float val = sc[tid][j];
                if (val > minv) {            // strict > keeps earlier index on ties
                    topV[tid][minslot] = val;
                    topI[tid][minslot] = nb + j;
                    minv = topV[tid][0]; minslot = 0;
#pragma unroll
                    for (int k = 1; k < K_; ++k) {
                        float tvk = topV[tid][k];
                        if (tvk < minv) { minv = tvk; minslot = k; }
                    }
                }
            }
        }
        __syncthreads();
    }

    if (tid < 16) {
        // selection sort: descending value, ties -> ascending index (top_k order)
        for (int k = 0; k < K_; ++k) {
            int best = k;
            for (int j = k + 1; j < K_; ++j) {
                float vj = topV[tid][j], vb = topV[tid][best];
                if (vj > vb || (vj == vb && topI[tid][j] < topI[tid][best])) best = j;
            }
            float tv = topV[tid][k]; topV[tid][k] = topV[tid][best]; topV[tid][best] = tv;
            int   ti = topI[tid][k]; topI[tid][k] = topI[tid][best]; topI[tid][best] = ti;
        }
        float mx = topV[tid][0];
        float e[K_];
        float sum = 0.0f;
#pragma unroll
        for (int k = 0; k < K_; ++k) { e[k] = __expf(topV[tid][k] - mx); sum += e[k]; }
        float inv = 1.0f / sum;
        int token = tok0 + tid;
#pragma unroll
        for (int k = 0; k < K_; ++k) {
            float w = e[k] * inv;
            int idx = topI[tid][k];
            tw[tid][k] = w;
            ow[(size_t)token * K_ + k]   = w;
            oidx[(size_t)token * K_ + k] = (float)idx;
            osel[(size_t)token * N_ + idx] = w;
        }
    }
    __syncthreads();

    // gather: output[token,:] = sum_k w_k * neurons[idx_k,:]
    for (int t = 0; t < 16; ++t) {
        int token = tok0 + t;
        for (int d = tid; d < D_; d += 256) {
            float accd = 0.0f;
#pragma unroll
            for (int k = 0; k < K_; ++k)
                accd += tw[t][k] * neurons[(size_t)topI[t][k] * D_ + d];
            outv[(size_t)token * D_ + d] = accd;
        }
    }
}

// ---------------------------------------------------------------------------
// Host launcher
// ---------------------------------------------------------------------------
extern "C" void kernel_launch(void* const* d_in, const int* in_sizes, int n_in,
                              void* d_out, int out_size, void* d_ws, size_t ws_size,
                              hipStream_t stream) {
    (void)in_sizes; (void)n_in; (void)out_size; (void)ws_size;

    const float* x       = (const float*)d_in[0];
    const float* neurons = (const float*)d_in[1];
    const float* q_w     = (const float*)d_in[2];
    const float* q_b     = (const float*)d_in[3];
    const float* k_w     = (const float*)d_in[4];
    const float* k_b     = (const float*)d_in[5];
    const float* v_w     = (const float*)d_in[6];
    const float* v_b     = (const float*)d_in[7];
    const float* path_w  = (const float*)d_in[8];
    const float* path_b  = (const float*)d_in[9];

    char* ws = (char*)d_ws;
    size_t off = 0;
    auto wsalloc = [&](size_t bytes) -> void* {
        void* p = ws + off;
        off += (bytes + 255) & ~(size_t)255;
        return p;
    };

    unsigned short* xbf  = (unsigned short*)wsalloc((size_t)M_ * D_ * 2);  //  8 MB
    unsigned short* nbf  = (unsigned short*)wsalloc((size_t)N_ * D_ * 2);  // 32 MB
    unsigned short* wqT  = (unsigned short*)wsalloc((size_t)D_ * D_ * 2);  //  2 MB
    unsigned short* wkT  = (unsigned short*)wsalloc((size_t)D_ * D_ * 2);
    unsigned short* wvT  = (unsigned short*)wsalloc((size_t)D_ * D_ * 2);
    unsigned short* qb16 = (unsigned short*)wsalloc((size_t)M_ * D_ * 2);  //  8 MB
    unsigned short* kb16 = (unsigned short*)wsalloc((size_t)M_ * D_ * 2);
    unsigned short* vt16 = (unsigned short*)wsalloc((size_t)M_ * D_ * 2);
    float*          ctxf = (float*)wsalloc((size_t)M_ * D_ * 4);           // 16 MB
    unsigned short* mbf  = (unsigned short*)wsalloc((size_t)M_ * D_ * 2);  //  8 MB

    // 1) precision conversion / weight transposition
    cvt_bf16_k<<<512, 256, 0, stream>>>(x, xbf, M_ * D_);
    cvt_bf16_k<<<2048, 256, 0, stream>>>(neurons, nbf, N_ * D_);
    cvt_tr_bf16_k<<<1024, 256, 0, stream>>>(q_w, wqT, D_, D_);
    cvt_tr_bf16_k<<<1024, 256, 0, stream>>>(k_w, wkT, D_, D_);
    cvt_tr_bf16_k<<<1024, 256, 0, stream>>>(v_w, wvT, D_, D_);

    // 2) QKV projections (WMMA bf16, 32x32 register-blocked)
    dim3 gq(D_ / 64, M_ / 128);
    qkv_gemm_k<<<gq, 256, 0, stream>>>(xbf, wqT, q_b, qb16, 0);
    qkv_gemm_k<<<gq, 256, 0, stream>>>(xbf, wkT, k_b, kb16, 0);
    qkv_gemm_k<<<gq, 256, 0, stream>>>(xbf, wvT, v_b, vt16, 1);

    // 3) attention (flash-style, WMMA bf16)
    attn_k<<<512, 256, 0, stream>>>(qb16, kb16, vt16, ctxf);

    // 4) path weights + mixed vector
    path_mix_k<<<M_, 256, 0, stream>>>(x, ctxf, path_w, path_b, mbf);

    // 5) outputs: [output | topk_idx | topk_weights | selection_out]
    float* outv = (float*)d_out;
    float* oidx = outv + (size_t)M_ * D_;
    float* ow   = oidx + (size_t)M_ * K_;
    float* osel = ow   + (size_t)M_ * K_;

    zero_k<<<2048, 256, 0, stream>>>(osel, (size_t)M_ * N_);
    score_topk_k<<<M_ / 16, 256, 0, stream>>>(mbf, nbf, neurons, outv, oidx, ow, osel);
}
